// BiLSTMClassifier_22574348108004
// MI455X (gfx1250) — matrix operand order repaired and checked by reading, not compiled, not run on hardware
//
#include <hip/hip_runtime.h>
#include <hip/hip_bf16.h>

// ---------------------------------------------------------------------------
// BiLSTM classifier for MI455X (gfx1250), bf16 WMMA, f32 accumulate.
// B=256, T=128, D=256, H=512 (LSTM1), H2=1024 (LSTM2), L=2.
// Round 2: software-pipelined K-loop (register double buffering) so weight/
// activation loads for step k+1 overlap the WMMAs of step k.
// ---------------------------------------------------------------------------

typedef __bf16 v16bf __attribute__((ext_vector_type(16)));
typedef __bf16 v8bf  __attribute__((ext_vector_type(8)));
typedef float  v8f   __attribute__((ext_vector_type(8)));

#define BB   256   // batch
#define TT   128   // time steps
#define DD   256   // input feature dim
#define HH   512   // LSTM1 hidden
#define HH2  1024  // LSTM2 hidden (= 2*HH)

__device__ __forceinline__ float sigf(float x)      { return 1.0f / (1.0f + __expf(-x)); }
__device__ __forceinline__ float tanh_fast(float x) { return 2.0f * sigf(2.0f * x) - 1.0f; }

// ---------------------------------------------------------------------------
// Elementwise conversion: f32 -> bf16
// ---------------------------------------------------------------------------
__global__ void f32_to_bf16_kernel(const float* __restrict__ in,
                                   __bf16* __restrict__ out, size_t n) {
    size_t i = (size_t)blockIdx.x * blockDim.x + threadIdx.x;
    if (i < n) out[i] = (__bf16)in[i];
}

// ---------------------------------------------------------------------------
// Build concatenated bf16 weight matrix: Wcat[n][k] =
//   k <  Din : Wih[n][k]
//   k >= Din : Whh[n][k-Din]
// Wcat shape [N4, Din+Hh], row major.
// ---------------------------------------------------------------------------
__global__ void build_wcat_kernel(const float* __restrict__ Wih,
                                  const float* __restrict__ Whh,
                                  __bf16* __restrict__ Wcat,
                                  int Din, int Hh, int N4) {
    size_t idx = (size_t)blockIdx.x * blockDim.x + threadIdx.x;
    size_t Kc = (size_t)Din + Hh;
    if (idx >= (size_t)N4 * Kc) return;
    int n = (int)(idx / Kc);
    int k = (int)(idx % Kc);
    float v = (k < Din) ? Wih[(size_t)n * Din + k]
                        : Whh[(size_t)n * Hh + (k - Din)];
    Wcat[idx] = (__bf16)v;
}

// ---------------------------------------------------------------------------
// Zero a region (4-byte words)
// ---------------------------------------------------------------------------
__global__ void zero_kernel(unsigned int* __restrict__ p, size_t nWords) {
    size_t i = (size_t)blockIdx.x * blockDim.x + threadIdx.x;
    if (i < nWords) p[i] = 0u;
}

// ---------------------------------------------------------------------------
// Fragment set for one K-step: A tile (16x32) + 4 B tiles (32x16 each)
// ---------------------------------------------------------------------------
struct Frags {
    v16bf a, b0, b1, b2, b3;
};

__device__ __forceinline__ void load_frags(Frags& f,
                                           const __bf16* __restrict__ xrow,
                                           const __bf16* __restrict__ hrow,
                                           const __bf16* __restrict__ wcol, // Wcat + (n0+r)*Kc + half*16
                                           int Kc, int Dx, int k0, int half) {
    // A fragment (16x32 bf16): lane holds K = half*16..+15
    const __bf16* asrc = (k0 < Dx) ? (xrow + k0) : (hrow + (k0 - Dx));
    union { v16bf v; v8bf h[2]; } af;
    af.h[0] = *(const v8bf*)(asrc + half * 16);
    af.h[1] = *(const v8bf*)(asrc + half * 16 + 8);
    f.a = af.v;
    // B fragments (32x16 bf16 each): lane holds col n0+r(+16t), K = k0+half*16..+15
    const __bf16* wbase = wcol + k0;
    f.b0 = *(const v16bf*)(wbase);
    f.b1 = *(const v16bf*)(wbase + (size_t)16 * Kc);
    f.b2 = *(const v16bf*)(wbase + (size_t)32 * Kc);
    f.b3 = *(const v16bf*)(wbase + (size_t)48 * Kc);
}

// ---------------------------------------------------------------------------
// Recurrent-step GEMM with bf16 WMMA:
//   R[b][n] = sum_k A[b][k] * Wcat[n][k],   A[b][:] = [ X[b][:Dx] | h[b][:Hh] ]
// X rows: Xpart + b*xRowStride (bf16). Each wave computes a 16(M) x 64(N) tile.
// K-loop is ping-pong pipelined: loads for step k+32 are issued before the
// WMMAs of step k, so memory latency overlaps matrix math.
// Requires Kc % 64 == 0 (holds: 768, 2048).
// ---------------------------------------------------------------------------
__global__ __launch_bounds__(128)
void wmma_gates_kernel(const __bf16* __restrict__ Xpart, size_t xRowStride,
                       const __bf16* __restrict__ Hpart, int Hh,
                       const __bf16* __restrict__ Wcat, int Kc, int Dx,
                       float* __restrict__ R, int N4) {
    const int lane   = threadIdx.x & 31;
    const int wave   = threadIdx.x >> 5;
    const int gw     = blockIdx.x * 4 + wave;
    const int ntiles = N4 >> 6;          // groups of 64 columns
    const int mt     = gw / ntiles;      // M tile (16 rows)
    const int ng     = gw % ntiles;      // N group (64 cols)
    const int m0     = mt * 16;
    const int n0     = ng * 64;
    const int r      = lane & 15;
    const int half   = lane >> 4;

    v8f acc0 = {}, acc1 = {}, acc2 = {}, acc3 = {};

    const int aRow = m0 + r;                              // batch row for this lane
    const __bf16* xrow = Xpart + (size_t)aRow * xRowStride;
    const __bf16* hrow = Hpart + (size_t)aRow * Hh;
    const __bf16* wcol = Wcat + (size_t)(n0 + r) * Kc + half * 16;

    Frags f0, f1;
    load_frags(f0, xrow, hrow, wcol, Kc, Dx, 0, half);

    for (int k0 = 0; k0 < Kc; k0 += 64) {
        // prefetch odd step while even step computes
        load_frags(f1, xrow, hrow, wcol, Kc, Dx, k0 + 32, half);
        acc0 = __builtin_amdgcn_wmma_f32_16x16x32_bf16(false, f0.a, false, f0.b0, (short)0, acc0, false, false);
        acc1 = __builtin_amdgcn_wmma_f32_16x16x32_bf16(false, f0.a, false, f0.b1, (short)0, acc1, false, false);
        acc2 = __builtin_amdgcn_wmma_f32_16x16x32_bf16(false, f0.a, false, f0.b2, (short)0, acc2, false, false);
        acc3 = __builtin_amdgcn_wmma_f32_16x16x32_bf16(false, f0.a, false, f0.b3, (short)0, acc3, false, false);
        // prefetch next even step while odd step computes
        if (k0 + 64 < Kc)
            load_frags(f0, xrow, hrow, wcol, Kc, Dx, k0 + 64, half);
        acc0 = __builtin_amdgcn_wmma_f32_16x16x32_bf16(false, f1.a, false, f1.b0, (short)0, acc0, false, false);
        acc1 = __builtin_amdgcn_wmma_f32_16x16x32_bf16(false, f1.a, false, f1.b1, (short)0, acc1, false, false);
        acc2 = __builtin_amdgcn_wmma_f32_16x16x32_bf16(false, f1.a, false, f1.b2, (short)0, acc2, false, false);
        acc3 = __builtin_amdgcn_wmma_f32_16x16x32_bf16(false, f1.a, false, f1.b3, (short)0, acc3, false, false);
    }

    // C layout: VGPR j -> row m0 + half*8 + j, col n0 + r (+16 per tile)
    float* out = R + (size_t)(m0 + half * 8) * N4 + n0 + r;
#pragma unroll
    for (int j = 0; j < 8; ++j) {
        out[(size_t)j * N4 +  0] = acc0[j];
        out[(size_t)j * N4 + 16] = acc1[j];
        out[(size_t)j * N4 + 32] = acc2[j];
        out[(size_t)j * N4 + 48] = acc3[j];
    }
}

// ---------------------------------------------------------------------------
// LSTM cell update (PyTorch gate order i,f,g,o):
//   c' = sig(f)*c + sig(i)*tanh(g);  h = sig(o)*tanh(c')
// ---------------------------------------------------------------------------
__global__ void lstm_cell_kernel(const float* __restrict__ R,
                                 const float* __restrict__ bias,
                                 float* __restrict__ c,
                                 __bf16* __restrict__ hbf,
                                 __bf16* __restrict__ outslot, int outStride,
                                 float* __restrict__ hf32, int Hh) {
    int idx = blockIdx.x * blockDim.x + threadIdx.x;   // 0 .. B*Hh-1
    int b = idx / Hh;
    int n = idx % Hh;
    const float* Rb = R + (size_t)b * (4 * (size_t)Hh);
    float gi = Rb[n]            + bias[n];
    float gf = Rb[Hh + n]       + bias[Hh + n];
    float gg = Rb[2 * Hh + n]   + bias[2 * Hh + n];
    float go = Rb[3 * Hh + n]   + bias[3 * Hh + n];
    float cn = sigf(gf) * c[idx] + sigf(gi) * tanh_fast(gg);
    float hv = sigf(go) * tanh_fast(cn);
    c[idx]   = cn;
    hbf[idx] = (__bf16)hv;
    if (outslot) outslot[(size_t)b * outStride + n] = (__bf16)hv;
    if (hf32)    hf32[idx] = hv;
}

// ---------------------------------------------------------------------------
// Final classifier: out[b][l] = sigmoid( dot(h2[b], Wl[l]) + bl[l] ), L=2
// ---------------------------------------------------------------------------
__global__ __launch_bounds__(128)
void classifier_kernel(const float* __restrict__ h2, const float* __restrict__ Wl,
                       const float* __restrict__ bl, float* __restrict__ out, int K) {
    __shared__ float s0[128];
    __shared__ float s1[128];
    int b = blockIdx.x;
    const float* hb = h2 + (size_t)b * K;
    float a0 = 0.f, a1 = 0.f;
    for (int k = threadIdx.x; k < K; k += 128) {
        float hv = hb[k];
        a0 += hv * Wl[k];
        a1 += hv * Wl[K + k];
    }
    s0[threadIdx.x] = a0;
    s1[threadIdx.x] = a1;
    __syncthreads();
    for (int off = 64; off > 0; off >>= 1) {
        if ((int)threadIdx.x < off) {
            s0[threadIdx.x] += s0[threadIdx.x + off];
            s1[threadIdx.x] += s1[threadIdx.x + off];
        }
        __syncthreads();
    }
    if (threadIdx.x == 0) {
        out[b * 2 + 0] = sigf(s0[0] + bl[0]);
        out[b * 2 + 1] = sigf(s1[0] + bl[1]);
    }
}

// ---------------------------------------------------------------------------
// Host-side orchestration
// ---------------------------------------------------------------------------
extern "C" void kernel_launch(void* const* d_in, const int* in_sizes, int n_in,
                              void* d_out, int out_size, void* d_ws, size_t ws_size,
                              hipStream_t stream) {
    (void)in_sizes; (void)n_in; (void)out_size; (void)ws_size;

    const float* x     = (const float*)d_in[0];
    const float* Wf_ih = (const float*)d_in[1];
    const float* Wf_hh = (const float*)d_in[2];
    const float* bf    = (const float*)d_in[3];
    const float* Wb_ih = (const float*)d_in[4];
    const float* Wb_hh = (const float*)d_in[5];
    const float* bb    = (const float*)d_in[6];
    const float* Ws_ih = (const float*)d_in[7];
    const float* Ws_hh = (const float*)d_in[8];
    const float* bs    = (const float*)d_in[9];
    const float* Wl    = (const float*)d_in[10];
    const float* bl    = (const float*)d_in[11];
    float* out = (float*)d_out;

    // ---- workspace layout (bytes, 256-aligned sections) ----
    constexpr size_t XBF_ELEMS   = (size_t)BB * TT * DD;               // 8.4M bf16
    constexpr size_t WFCAT_ELEMS = (size_t)(4 * HH) * (DD + HH);       // 2048x768
    constexpr size_t WSCAT_ELEMS = (size_t)(4 * HH2) * (2 * HH + HH2); // 4096x2048
    constexpr size_t COMB_ELEMS  = (size_t)TT * BB * (2 * HH);         // 33.5M bf16
    constexpr size_t HF_ELEMS    = (size_t)BB * HH;
    constexpr size_t H2_ELEMS    = (size_t)BB * HH2;
    constexpr size_t R_ELEMS     = (size_t)BB * (4 * HH2);             // covers both LSTMs

    char* p = (char*)d_ws;
    size_t off = 0;
    auto take = [&](size_t bytes) { char* q = p + off; off = (off + bytes + 255) & ~(size_t)255; return q; };

    __bf16* x_bf   = (__bf16*)take(XBF_ELEMS * 2);
    __bf16* wf_cat = (__bf16*)take(WFCAT_ELEMS * 2);
    __bf16* wb_cat = (__bf16*)take(WFCAT_ELEMS * 2);
    __bf16* ws_cat = (__bf16*)take(WSCAT_ELEMS * 2);
    __bf16* comb   = (__bf16*)take(COMB_ELEMS * 2);
    float*  Rbuf   = (float*)take(R_ELEMS * 4);
    // contiguous zero-init state region:
    char*   stateBase = p + off;
    __bf16* hf_bf  = (__bf16*)take(HF_ELEMS * 2);
    float*  cf     = (float*)take(HF_ELEMS * 4);
    __bf16* hb_bf  = (__bf16*)take(HF_ELEMS * 2);
    float*  cb     = (float*)take(HF_ELEMS * 4);
    __bf16* h2_bf  = (__bf16*)take(H2_ELEMS * 2);
    float*  c2     = (float*)take(H2_ELEMS * 4);
    float*  h2f    = (float*)take(H2_ELEMS * 4);
    size_t stateBytes = (size_t)((p + off) - stateBase);

    // ---- conversions / weight packing ----
    {
        int thr = 256;
        f32_to_bf16_kernel<<<(int)((XBF_ELEMS + thr - 1) / thr), thr, 0, stream>>>(x, x_bf, XBF_ELEMS);
        build_wcat_kernel<<<(int)((WFCAT_ELEMS + thr - 1) / thr), thr, 0, stream>>>(Wf_ih, Wf_hh, wf_cat, DD, HH, 4 * HH);
        build_wcat_kernel<<<(int)((WFCAT_ELEMS + thr - 1) / thr), thr, 0, stream>>>(Wb_ih, Wb_hh, wb_cat, DD, HH, 4 * HH);
        build_wcat_kernel<<<(int)((WSCAT_ELEMS + thr - 1) / thr), thr, 0, stream>>>(Ws_ih, Ws_hh, ws_cat, 2 * HH, HH2, 4 * HH2);
        size_t words = stateBytes / 4;
        zero_kernel<<<(int)((words + thr - 1) / thr), thr, 0, stream>>>((unsigned int*)stateBase, words);
    }

    const int N4_1 = 4 * HH;       // 2048
    const int N4_2 = 4 * HH2;      // 4096
    const int Kc1  = DD + HH;      // 768
    const int Kc2  = 2 * HH + HH2; // 2048
    const int gemmBlocks1 = (16 * (N4_1 / 64)) / 4;   // 128 blocks * 4 waves
    const int gemmBlocks2 = (16 * (N4_2 / 64)) / 4;   // 256
    const int cellBlocks1 = (BB * HH) / 256;          // 512
    const int cellBlocks2 = (BB * HH2) / 256;         // 1024

    // ---- forward LSTM over t = 0..T-1; h -> combined[t][:, 0:H] ----
    for (int t = 0; t < TT; ++t) {
        wmma_gates_kernel<<<gemmBlocks1, 128, 0, stream>>>(
            x_bf + (size_t)t * DD, (size_t)TT * DD, hf_bf, HH,
            wf_cat, Kc1, DD, Rbuf, N4_1);
        lstm_cell_kernel<<<cellBlocks1, 256, 0, stream>>>(
            Rbuf, bf, cf, hf_bf,
            comb + (size_t)t * BB * (2 * HH), 2 * HH, nullptr, HH);
    }

    // ---- backward LSTM, append order i=0..T-1 reads x[T-1-i]; h -> combined[i][:, H:2H]
    for (int i = 0; i < TT; ++i) {
        int t = TT - 1 - i;
        wmma_gates_kernel<<<gemmBlocks1, 128, 0, stream>>>(
            x_bf + (size_t)t * DD, (size_t)TT * DD, hb_bf, HH,
            wb_cat, Kc1, DD, Rbuf, N4_1);
        lstm_cell_kernel<<<cellBlocks1, 256, 0, stream>>>(
            Rbuf, bb, cb, hb_bf,
            comb + (size_t)i * BB * (2 * HH) + HH, 2 * HH, nullptr, HH);
    }

    // ---- second LSTM over combined sequence; keep final hidden (f32) ----
    for (int t = 0; t < TT; ++t) {
        wmma_gates_kernel<<<gemmBlocks2, 128, 0, stream>>>(
            comb + (size_t)t * BB * (2 * HH), (size_t)(2 * HH), h2_bf, HH2,
            ws_cat, Kc2, 2 * HH, Rbuf, N4_2);
        lstm_cell_kernel<<<cellBlocks2, 256, 0, stream>>>(
            Rbuf, bs, c2, h2_bf, nullptr, 0, h2f, HH2);
    }

    // ---- classifier ----
    classifier_kernel<<<BB, 128, 0, stream>>>(h2f, Wl, bl, out, 2 * HH);
}
